// PixelAGG_N_31490700214894
// MI455X (gfx1250) — matrix operand order repaired and checked by reading, not compiled, not run on hardware
//
#include <hip/hip_runtime.h>
#include <hip/hip_bf16.h>

// ---------------------------------------------------------------------------
// Non-local block (embedded gaussian attention), MI455X / gfx1250.
// Compute-bound (~51 GFLOP flash-recompute attention vs ~20MB traffic), so
// all heavy math runs on v_wmma_f32_16x16x32_bf16 (deepest bf16 MAC/issue).
// S (256MB fp32) is never materialized: pass A = online row logsumexp,
// pass B = recompute S tiles, exp2 in-register, feed AV WMMAs.
// Scheduling: batched fragment loads ahead of WMMA chains + double-buffered
// LDS staging with one barrier per tile to hide load latency.
// ---------------------------------------------------------------------------

typedef __bf16 bf16;
typedef bf16  bf16x8 __attribute__((ext_vector_type(8)));
typedef bf16  v16bf  __attribute__((ext_vector_type(16)));
typedef float v8f    __attribute__((ext_vector_type(8)));
typedef float f32x4  __attribute__((ext_vector_type(4)));

#define NTOK 8192   // n*h*w
#define CP   128    // inter channels == reshaped "channel" rows
#define CIN  256
#define KHW  1024   // h*w
// fold softmax scale 1/sqrt(128) and log2(e) into theta so exp() == v_exp_f32
#define THETA_SCALE (1.44269504088896340736f * 0.08838834764831844055f)

#if defined(__HIP_DEVICE_COMPILE__) && __has_builtin(__builtin_amdgcn_exp2f)
#  define EXP2F(x) __builtin_amdgcn_exp2f(x)
#else
#  define EXP2F(x) exp2f(x)
#endif
#if defined(__HIP_DEVICE_COMPILE__) && __has_builtin(__builtin_amdgcn_logf)
#  define LOG2F(x) __builtin_amdgcn_logf(x)
#else
#  define LOG2F(x) log2f(x)
#endif

static __device__ __forceinline__ v16bf make_frag(bf16x8 lo, bf16x8 hi) {
  union { v16bf v; bf16x8 h[2]; } u;
  u.h[0] = lo; u.h[1] = hi;
  return u.v;
}
static __device__ __forceinline__ bf16x8 ldg8(const bf16* p) {
  return *(const bf16x8*)p;
}
static __device__ __forceinline__ v8f vzero8() {
  v8f v;
#pragma unroll
  for (int i = 0; i < 8; ++i) v[i] = 0.0f;
  return v;
}
static __device__ __forceinline__ v8f wmma_bf16(v16bf a, v16bf b, v8f c) {
  // D = A(16x32) * B(32x16) + C, f32 accum
  return __builtin_amdgcn_wmma_f32_16x16x32_bf16(false, a, false, b,
                                                 (short)0, c, false, false);
}

// ---------------------------------------------------------------------------
// Kernel 1: three 1x1 projections. Writes bf16 matrices in the *reshaped*
// coordinate system of the reference's raw .view:
//   a = n*16 + c'/8        (matrix row, 0..127)
//   b = (c'%8)*1024 + hw   (matrix col / token, 0..8191)
// Layouts chosen for WMMA fragment gathers:
//   Tt[b*128 + a]  (theta, pre-scaled, row-per-token -> A frags & LDS tiles)
//   Pt[b*128 + a]  (phi, row-per-token -> B frags)
//   G [a*8192 + b] (g, row-per-channel -> A frags of the AV matmul)
// ---------------------------------------------------------------------------
__global__ __launch_bounds__(256) void nlb_proj(
    const float* __restrict__ x,
    const float* __restrict__ Wg, const float* __restrict__ bg,
    const float* __restrict__ Wt, const float* __restrict__ bt,
    const float* __restrict__ Wp, const float* __restrict__ bp,
    bf16* __restrict__ Tt, bf16* __restrict__ Pt, bf16* __restrict__ G) {
  __shared__ float xs[CIN][32];
  const int tid = threadIdx.x;
  const int t0  = blockIdx.x * 32;       // 32 natural tokens (same frame n)
  const int n   = t0 >> 10;
  const int hw0 = t0 & (KHW - 1);
  const float* xb = x + (size_t)n * CIN * KHW + hw0;

#pragma unroll 4
  for (int idx = tid; idx < CIN * 32; idx += 256)
    xs[idx >> 5][idx & 31] = xb[(idx >> 5) * KHW + (idx & 31)];
  __syncthreads();

  const int o  = tid & 127;   // output channel c'
  const int th = tid >> 7;    // token half (16 tokens each)
  float ag[16], at[16], ap[16];
#pragma unroll
  for (int k = 0; k < 16; ++k) { ag[k] = 0.f; at[k] = 0.f; ap[k] = 0.f; }

  for (int c = 0; c < CIN; ++c) {
    const float wg = Wg[o * CIN + c];
    const float wt = Wt[o * CIN + c];
    const float wp = Wp[o * CIN + c];
    const float* xr = &xs[c][th * 16];
#pragma unroll
    for (int k = 0; k < 16; ++k) {
      const float xv = xr[k];
      ag[k] = fmaf(wg, xv, ag[k]);
      at[k] = fmaf(wt, xv, at[k]);
      ap[k] = fmaf(wp, xv, ap[k]);
    }
  }
  const float bgv = bg[o], btv = bt[o], bpv = bp[o];
  const int a = (n << 4) + (o >> 3);
  const int bbase = (o & 7) << 10;
#pragma unroll
  for (int k = 0; k < 16; ++k) {
    const int hw = hw0 + th * 16 + k;
    const int b  = bbase + hw;
    Tt[(size_t)b * CP + a] = (bf16)((at[k] + btv) * THETA_SCALE);
    Pt[(size_t)b * CP + a] = (bf16)(ap[k] + bpv);
    G [(size_t)a * NTOK + b] = (bf16)(ag[k] + bgv);
  }
}

// ---------------------------------------------------------------------------
// Kernel 2 (pass A): per-row logsumexp of S = theta^T * phi (base-2 domain).
// 128 threads = 4 waves, each wave owns 16 rows j; theta A-frags resident in
// VGPRs. Phi staged in double-buffered LDS (32 cols / step, one barrier),
// fragment loads batched ahead of the 8 WMMAs per step.
// ---------------------------------------------------------------------------
__global__ __launch_bounds__(128) void nlb_lse(
    const bf16* __restrict__ Tt, const bf16* __restrict__ Pt,
    float* __restrict__ L) {
  __shared__ __align__(16) bf16 pbuf[2][32 * CP];   // 2 x 8KB phi tile [i][c]
  const int tid  = threadIdx.x;
  const int lane = tid & 31, wv = tid >> 5;
  const int lmod = lane & 15, lhi = lane >> 4;
  const int j0   = blockIdx.x * 64 + wv * 16;

  // theta A fragments: rows j0..j0+15, K = c (128) -> 4 frags
  v16bf afrag[4];
  {
    const bf16* tr = Tt + (size_t)(j0 + lmod) * CP;
#pragma unroll
    for (int kb = 0; kb < 4; ++kb) {
      const int c0 = kb * 32 + 8 * lhi;
      afrag[kb] = make_frag(ldg8(tr + c0), ldg8(tr + c0 + 16));
    }
  }

  // cooperative staging: thread covers 64B of a 32x128 tile
  const int srow = tid >> 2;          // 0..31
  const int scol = (tid & 3) * 32;    // 0,32,64,96
  bf16x8 st[4];
  {
    const bf16* src = Pt + (size_t)srow * CP + scol;
#pragma unroll
    for (int q = 0; q < 4; ++q) st[q] = ldg8(src + 8 * q);
  }

  float m[8], s[8];
#pragma unroll
  for (int k = 0; k < 8; ++k) { m[k] = -3.0e38f; s[k] = 0.f; }

  int p = 0;
  for (int ib = 0; ib < NTOK / 32; ++ib) {
    {  // publish staged tile, one barrier, then immediately fetch next
      bf16* dst = &pbuf[p][srow * CP + scol];
#pragma unroll
      for (int q = 0; q < 4; ++q) *(bf16x8*)(dst + 8 * q) = st[q];
    }
    __syncthreads();
    if (ib + 1 < NTOK / 32) {
      const bf16* src = Pt + (size_t)((ib + 1) * 32 + srow) * CP + scol;
#pragma unroll
      for (int q = 0; q < 4; ++q) st[q] = ldg8(src + 8 * q);
      __builtin_prefetch(src + 64 * CP, 0, 1);   // global_prefetch_b8
    }

#pragma unroll
    for (int half = 0; half < 2; ++half) {
      // batch all 4 B fragments, then issue the WMMA chain back-to-back
      v16bf bfrg[4];
#pragma unroll
      for (int kb = 0; kb < 4; ++kb) {
        const bf16* pp = &pbuf[p][(half * 16 + lmod) * CP + kb * 32 + 8 * lhi];
        bfrg[kb] = make_frag(*(const bf16x8*)pp, *(const bf16x8*)(pp + 16));
      }
      v8f acc = vzero8();
#pragma unroll
      for (int kb = 0; kb < 4; ++kb) acc = wmma_bf16(afrag[kb], bfrg[kb], acc);

      // online logsumexp update (base-2; scale folded into theta)
#pragma unroll
      for (int k = 0; k < 8; ++k) {
        const float xv = acc[k];
        const float nm = fmaxf(m[k], xv);
        s[k] = s[k] * EXP2F(m[k] - nm) + EXP2F(xv - nm);
        m[k] = nm;
      }
    }
    p ^= 1;
  }

  // reduce across the 16 lanes of each half-wave (D-tile columns)
#pragma unroll
  for (int off = 1; off < 16; off <<= 1) {
#pragma unroll
    for (int k = 0; k < 8; ++k) {
      const float mo = __shfl_xor(m[k], off, 32);
      const float so = __shfl_xor(s[k], off, 32);
      const float nm = fmaxf(m[k], mo);
      s[k] = s[k] * EXP2F(m[k] - nm) + so * EXP2F(mo - nm);
      m[k] = nm;
    }
  }
  if (lmod == 0) {
#pragma unroll
    for (int k = 0; k < 8; ++k)
      L[j0 + lhi * 8 + k] = m[k] + LOG2F(s[k]);
  }
}

// ---------------------------------------------------------------------------
// Kernel 3 (pass B): Y = G * exp2(S - L[j])  (flash-style recompute).
// 128 threads = 4 waves; wave owns 16 output columns i, all 128 rows of Y.
// Per 32-row j-step: theta tile via double-buffered LDS (one barrier),
// batched A-frag loads -> 8 WMMAs for S, exp2 against logsumexp, bf16 repack
// through per-wave LDS transpose into the E B-fragment, G A-frag global loads
// interleaved in batches of 4 with the 8 AV WMMAs.
// ---------------------------------------------------------------------------
__global__ __launch_bounds__(128) void nlb_attn(
    const bf16* __restrict__ Tt, const bf16* __restrict__ Pt,
    const bf16* __restrict__ G,  const float* __restrict__ L,
    float* __restrict__ Y) {
  __shared__ __align__(16) bf16 tbuf[2][32 * CP];  // 2 x 8KB theta tile [j][c]
  __shared__ __align__(16) bf16 ebuf[4][16 * 32];  // per-wave E tile [i][j]
  const int tid  = threadIdx.x;
  const int lane = tid & 31, wv = tid >> 5;
  const int lmod = lane & 15, lhi = lane >> 4;
  const int i0   = blockIdx.x * 64 + wv * 16;
  bf16* eb = &ebuf[wv][0];

  // phi B fragments for our 16 columns (fixed for whole kernel)
  v16bf pf[4];
  {
    const bf16* pr = Pt + (size_t)(i0 + lmod) * CP;
#pragma unroll
    for (int kb = 0; kb < 4; ++kb) {
      const int c0 = kb * 32 + 8 * lhi;
      pf[kb] = make_frag(ldg8(pr + c0), ldg8(pr + c0 + 16));
    }
  }

  // cooperative staging of theta tiles (thread covers 64B of 32x128)
  const int srow = tid >> 2;
  const int scol = (tid & 3) * 32;
  bf16x8 st[4];
  {
    const bf16* src = Tt + (size_t)srow * CP + scol;
#pragma unroll
    for (int q = 0; q < 4; ++q) st[q] = ldg8(src + 8 * q);
  }

  v8f yacc[8];
#pragma unroll
  for (int ct = 0; ct < 8; ++ct) yacc[ct] = vzero8();

  int p = 0;
  for (int jb = 0; jb < NTOK / 32; ++jb) {
    const int j0 = jb * 32;
    {
      bf16* dst = &tbuf[p][srow * CP + scol];
#pragma unroll
      for (int q = 0; q < 4; ++q) *(bf16x8*)(dst + 8 * q) = st[q];
    }
    __syncthreads();
    if (jb + 1 < NTOK / 32) {
      const bf16* src = Tt + (size_t)(j0 + 32 + srow) * CP + scol;
#pragma unroll
      for (int q = 0; q < 4; ++q) st[q] = ldg8(src + 8 * q);
      __builtin_prefetch(src + 64 * CP, 0, 1);   // global_prefetch_b8
    }

    // --- S tiles for j0..j0+31, exp2, pack to LDS in [i][j] order ---
#pragma unroll
    for (int h = 0; h < 2; ++h) {
      v16bf af[4];
#pragma unroll
      for (int kb = 0; kb < 4; ++kb) {
        const bf16* tp = &tbuf[p][(h * 16 + lmod) * CP + kb * 32 + 8 * lhi];
        af[kb] = make_frag(*(const bf16x8*)tp, *(const bf16x8*)(tp + 16));
      }
      v8f sacc = vzero8();
#pragma unroll
      for (int kb = 0; kb < 4; ++kb) sacc = wmma_bf16(af[kb], pf[kb], sacc);

      const float* lp = L + j0 + h * 16 + 8 * lhi;
      f32x4 l0 = *(const f32x4*)lp;
      f32x4 l1 = *(const f32x4*)(lp + 4);
      bf16x8 ev;
#pragma unroll
      for (int k = 0; k < 8; ++k) {
        const float lk = (k < 4) ? l0[k] : l1[k - 4];
        ev[k] = (bf16)EXP2F(sacc[k] - lk);
      }
      // D-tile (j,i) -> [i][j] LDS; same-wave DS ops are in-order
      *(bf16x8*)&eb[lmod * 32 + h * 16 + 8 * lhi] = ev;
    }

    // --- AV: Y += G[:, j0..j0+31] * E ---  (G loads batched by 4)
    v16bf gf[4];
#pragma unroll
    for (int ct = 0; ct < 4; ++ct) {   // issue first batch early (L2 latency)
      const bf16* gp = G + (size_t)(ct * 16 + lmod) * NTOK + j0 + 8 * lhi;
      gf[ct] = make_frag(ldg8(gp), ldg8(gp + 16));
    }
    const bf16* ep = &eb[lmod * 32 + 8 * lhi];
    v16bf efrag = make_frag(*(const bf16x8*)ep, *(const bf16x8*)(ep + 16));
#pragma unroll
    for (int ct = 0; ct < 4; ++ct) yacc[ct] = wmma_bf16(gf[ct], efrag, yacc[ct]);
#pragma unroll
    for (int ct = 4; ct < 8; ++ct) {
      const bf16* gp = G + (size_t)(ct * 16 + lmod) * NTOK + j0 + 8 * lhi;
      gf[ct - 4] = make_frag(ldg8(gp), ldg8(gp + 16));
    }
#pragma unroll
    for (int ct = 4; ct < 8; ++ct)
      yacc[ct] = wmma_bf16(gf[ct - 4], efrag, yacc[ct]);

    p ^= 1;
  }

  // store Y (matrix coords) as f32
#pragma unroll
  for (int ct = 0; ct < 8; ++ct) {
#pragma unroll
    for (int k = 0; k < 8; ++k)
      Y[(size_t)(ct * 16 + 8 * lhi + k) * NTOK + i0 + lmod] = yacc[ct][k];
  }
}

// ---------------------------------------------------------------------------
// Kernel 4: out = x + Wo @ y + bo, mapping Y back out of the raw-view coords.
// ---------------------------------------------------------------------------
__global__ __launch_bounds__(256) void nlb_out(
    const float* __restrict__ x, const float* __restrict__ Y,
    const float* __restrict__ Wo, const float* __restrict__ bo,
    float* __restrict__ out) {
  __shared__ float ys[CP][64];   // 32KB: y[n, c', hw0..hw0+63]
  const int tid = threadIdx.x;
  const int n   = blockIdx.x >> 4;
  const int hw0 = (blockIdx.x & 15) * 64;

#pragma unroll 4
  for (int idx = tid; idx < CP * 64; idx += 256) {
    const int cp = idx >> 6, tt = idx & 63;
    ys[cp][tt] = Y[(size_t)((n << 4) + (cp >> 3)) * NTOK +
                   ((cp & 7) << 10) + hw0 + tt];
  }
  __syncthreads();

  const int c = tid;                 // 256 output channels
  const float* wrow = Wo + c * CP;
  const float bov = bo[c];
  for (int q = 0; q < 4; ++q) {
    float acc[16];
#pragma unroll
    for (int k = 0; k < 16; ++k) acc[k] = 0.f;
    for (int cp = 0; cp < CP; ++cp) {
      const float w = wrow[cp];
      const float* yr = &ys[cp][q * 16];
#pragma unroll
      for (int k = 0; k < 16; ++k) acc[k] = fmaf(w, yr[k], acc[k]);
    }
    const size_t base = (size_t)n * CIN * KHW + (size_t)c * KHW + hw0 + q * 16;
#pragma unroll
    for (int k = 0; k < 16; ++k) out[base + k] = x[base + k] + acc[k] + bov;
  }
}

// ---------------------------------------------------------------------------
extern "C" void kernel_launch(void* const* d_in, const int* in_sizes, int n_in,
                              void* d_out, int out_size, void* d_ws,
                              size_t ws_size, hipStream_t stream) {
  (void)in_sizes; (void)n_in; (void)out_size; (void)ws_size;
  const float* x  = (const float*)d_in[0];
  const float* Wg = (const float*)d_in[1];
  const float* bg = (const float*)d_in[2];
  const float* Wt = (const float*)d_in[3];
  const float* bt = (const float*)d_in[4];
  const float* Wp = (const float*)d_in[5];
  const float* bp = (const float*)d_in[6];
  const float* Wo = (const float*)d_in[7];
  const float* bo = (const float*)d_in[8];
  float* out = (float*)d_out;

  // workspace layout (bytes): Tt 2MB | Pt 2MB | G 2MB | L 64KB | Y 4MB
  char* ws = (char*)d_ws;
  bf16*  Tt = (bf16*)(ws);
  bf16*  Pt = (bf16*)(ws + (size_t)2 * 1024 * 1024);
  bf16*  G  = (bf16*)(ws + (size_t)4 * 1024 * 1024);
  float* L  = (float*)(ws + (size_t)6 * 1024 * 1024);
  float* Y  = (float*)(ws + (size_t)6 * 1024 * 1024 + 64 * 1024);

  nlb_proj<<<NTOK / 32, 256, 0, stream>>>(x, Wg, bg, Wt, bt, Wp, bp, Tt, Pt, G);
  nlb_lse <<<NTOK / 64, 128, 0, stream>>>(Tt, Pt, L);
  nlb_attn<<<NTOK / 64, 128, 0, stream>>>(Tt, Pt, G, L, Y);
  nlb_out <<<NTOK / 64, 256, 0, stream>>>(x, Y, Wo, bo, out);
}
